// MDI_2422361555492
// MI455X (gfx1250) — matrix operand order repaired and checked by reading, not compiled, not run on hardware
//
#include <hip/hip_runtime.h>

// MI455X fused kernel: bf16 WMMA GEMMs; weights pre-converted to transposed bf16
// once (pre-kernel), then streamed global->LDS with the Tensor Data Mover,
// double-buffered so TDM overlaps WMMA. Activations stream once from HBM
// (memory-bound floor ~11.5us @ 23.3 TB/s).
#define C    128
#define TB   256
#define ROWS 128
#define FEPS 1e-8f

typedef __bf16 bf16;
typedef __attribute__((ext_vector_type(16))) __bf16 v16bf;
typedef __attribute__((ext_vector_type(8)))  __bf16 v8bf;
typedef __attribute__((ext_vector_type(4)))  __bf16 v4bf;
typedef __attribute__((ext_vector_type(8)))  float  v8f;
typedef __attribute__((ext_vector_type(4)))  float  v4f;
typedef __attribute__((ext_vector_type(4)))  unsigned int v4u;
typedef __attribute__((ext_vector_type(8)))  int    v8i;
typedef __attribute__((ext_vector_type(4)))  int    v4i;

// ---------------- TDM: 128x128 bf16 tile, global -> LDS ----------------
__device__ __forceinline__ unsigned lds_addr_of(const void* p) {
  // generic -> LDS addrspace(3) -> 32-bit LDS byte offset
  return (unsigned)(unsigned long long)(__attribute__((address_space(3))) const char*)p;
}

__device__ __forceinline__ void tdm_load_weight(const bf16* gsrc, bf16* ldst) {
  unsigned lo = lds_addr_of(ldst);
  unsigned long long ga = (unsigned long long)gsrc;
  // D# group0: count=1 | lds_addr | global_addr[56:0] | type=2 (bits 127:126)
  v4u g0 = { 1u, lo, (unsigned)ga,
             (unsigned)((ga >> 32) & 0x1FFFFFFull) | 0x80000000u };
  // D# group1: data_size=2B; tensor 128x128, tile 128x128, stride0=128 elems
  v8i g1 = { 0x00010000,   // data_size=1 (2 bytes) at bits 17:16
             0x00800000,   // tensor_dim0 low16 = 128 at bits 63:48
             0x00800000,   // tensor_dim1 low16 = 128 at bits 95:80
             0x00800000,   // tile_dim0 = 128 at bits 127:112
             0x00000080,   // tile_dim1 = 128 at bits 143:128
             0x00000080,   // tensor_dim0_stride = 128 at bits 191:160
             0, 0 };
  v4i gz4 = { 0, 0, 0, 0 };
  v8i gz8 = { 0, 0, 0, 0, 0, 0, 0, 0 };
  // 6-arg form (clang-23 / therock HIP headers)
  __builtin_amdgcn_tensor_load_to_lds(g0, g1, gz4, gz4, gz8, 0);
}

// ---------------- Fallback: stage fp32 W[k][n] -> LDS bf16 Wt[n][k] ----------------
__device__ __forceinline__ void stage_weight(const float* __restrict__ W,
                                             bf16* __restrict__ dst, int tid) {
#pragma unroll
  for (int i = 0; i < 16; ++i) {
    int idx = i * 1024 + tid * 4;
    int k = idx >> 7;
    int n = idx & 127;
    v4f w = *(const v4f*)(W + idx);
    dst[(n + 0) * C + k] = (bf16)w.x;
    dst[(n + 1) * C + k] = (bf16)w.y;
    dst[(n + 2) * C + k] = (bf16)w.z;
    dst[(n + 3) * C + k] = (bf16)w.w;
  }
}

// Issue staging of one weight into `slot` (TDM from bf16 ws, or manual from fp32).
template <bool ASYNCW>
__device__ __forceinline__ void issue_w(const bf16* src_bf, const float* src_f,
                                        bf16* slot, int tid, int wave) {
  if constexpr (ASYNCW) {
    if (wave == 0) tdm_load_weight(src_bf, slot);   // one TDM per WG, async
  } else {
    stage_weight(src_f, slot, tid);
  }
}

template <bool ASYNCW>
__device__ __forceinline__ void wait_w(int wave) {
  if constexpr (ASYNCW) {
    if (wave == 0) __builtin_amdgcn_s_wait_tensorcnt(0);
  }
  __syncthreads();
}

// ---------------- WMMA fragment helpers ----------------
__device__ __forceinline__ v16bf load_afrag(const bf16* act, int r0, int k0, int lh, int hi) {
  const bf16* p = act + (r0 + lh) * C + k0 + hi * 16;
  v8bf lo = *(const v8bf*)(p);
  v8bf hb = *(const v8bf*)(p + 8);
  return __builtin_shufflevector(lo, hb, 0,1,2,3,4,5,6,7,8,9,10,11,12,13,14,15);
}

__device__ __forceinline__ v16bf load_bfrag(const bf16* wt, int n0, int k0, int lh, int hi) {
  return *(const v16bf*)(wt + (n0 + lh) * C + k0 + hi * 16);
}

__device__ __forceinline__ v8f gemm_tile(const bf16* act, const bf16* wt,
                                         int r0, int n0, int lh, int hi) {
  v8f acc = {};
#pragma unroll
  for (int k0 = 0; k0 < C; k0 += 32) {
    v16bf a = load_afrag(act, r0, k0, lh, hi);
    v16bf b = load_bfrag(wt, n0, k0, lh, hi);
    acc = __builtin_amdgcn_wmma_f32_16x16x32_bf16(false, a, false, b,
                                                  (short)0, acc, false, false);
  }
  return acc;
}

__device__ __forceinline__ void store_cfrag(bf16* act, int r0, int n0, int lh, int hi, v8f c) {
#pragma unroll
  for (int j = 0; j < 8; ++j)
    act[(r0 + hi * 8 + j) * C + n0 + lh] = (bf16)c[j];
}

__device__ __forceinline__ v8f wave_reduce16(v8f x) {
#pragma unroll
  for (int j = 0; j < 8; ++j) {
    float v = x[j];
    v += __shfl_xor(v, 1, 32);
    v += __shfl_xor(v, 2, 32);
    v += __shfl_xor(v, 4, 32);
    v += __shfl_xor(v, 8, 32);
    x[j] = v;
  }
  return x;
}

// ---------------- Layer attention over 2 layers, query from x1 ----------------
// Pattern per sub-phase: [issue next weight] [WMMA with current slot] [wait].
template <bool FUSE, bool ASYNCW>
__device__ __forceinline__ void layer_att(const bf16* x0, const bf16* x1, bf16* nebuf,
                                          const bf16* slotQ, const bf16* slotK, bf16* slotV,
                                          const bf16* wkb, const float* wkf,
                                          const bf16* wvb, const float* wvf,
                                          const bf16* nxb, const float* nxf, bf16* nxslot,
                                          int tid, int wave, int r0, int lh, int hi) {
  issue_w<ASYNCW>(wkb, wkf, (bf16*)slotK, tid, wave);
  v8f q[8];
#pragma unroll
  for (int t = 0; t < 8; ++t) q[t] = gemm_tile(x1, slotQ, r0, t * 16, lh, hi);
  wait_w<ASYNCW>(wave);

  issue_w<ASYNCW>(wvb, wvf, slotV, tid, wave);
  v8f s0 = {}, s1 = {};
#pragma unroll
  for (int t = 0; t < 8; ++t) {
    v8f k0 = gemm_tile(x0, slotK, r0, t * 16, lh, hi);
    v8f k1 = gemm_tile(x1, slotK, r0, t * 16, lh, hi);
#pragma unroll
    for (int j = 0; j < 8; ++j) {
      s0[j] += q[t][j] * k0[j];
      s1[j] += q[t][j] * k1[j];
    }
  }
  s0 = wave_reduce16(s0);
  s1 = wave_reduce16(s1);
  float a0[8], a1[8];
#pragma unroll
  for (int j = 0; j < 8; ++j) {
    float t0 = s0[j] * 0.08838834764831845f;  // 1/sqrt(128)
    float t1 = s1[j] * 0.08838834764831845f;
    float mx = fmaxf(t0, t1);
    float e0 = __expf(t0 - mx);
    float e1 = __expf(t1 - mx);
    float inv = 1.f / (e0 + e1);
    a0[j] = e0 * inv;
    a1[j] = e1 * inv;
  }
  wait_w<ASYNCW>(wave);

  issue_w<ASYNCW>(nxb, nxf, nxslot, tid, wave);
  if (!FUSE) {
    v8f acc[8];
#pragma unroll
    for (int t = 0; t < 8; ++t) {
      v8f v0 = gemm_tile(x0, slotV, r0, t * 16, lh, hi);
      v8f v1 = gemm_tile(x1, slotV, r0, t * 16, lh, hi);
#pragma unroll
      for (int j = 0; j < 8; ++j) acc[t][j] = a0[j] * v0[j] + a1[j] * v1[j];
    }
    // nebuf may alias x0: this wave's x0 reads precede these stores (DS in-order),
    // and waves touch disjoint row blocks.
#pragma unroll
    for (int t = 0; t < 8; ++t) store_cfrag(nebuf, r0, t * 16, lh, hi, acc[t]);
  } else {
#pragma unroll
    for (int t = 0; t < 8; ++t) {
      v8f v0 = gemm_tile(x0, slotV, r0, t * 16, lh, hi);
      v8f v1 = gemm_tile(x1, slotV, r0, t * 16, lh, hi);
#pragma unroll
      for (int j = 0; j < 8; ++j) {
        float dl = a0[j] * v0[j] + a1[j] * v1[j];
        int idx = (r0 + hi * 8 + j) * C + t * 16 + lh;
        nebuf[idx] = (bf16)((float)nebuf[idx] * dl);
      }
    }
  }
  wait_w<ASYNCW>(wave);
}

// ---------------- Pre-kernel: fp32 W[k][n] -> transposed bf16 ws[w][n][k] ----------------
extern "C" __global__ void __launch_bounds__(256)
cvt_weights_kernel(const float* __restrict__ Wg,  const float* __restrict__ Wqm,
                   const float* __restrict__ Wkm, const float* __restrict__ Wvm,
                   const float* __restrict__ Wqd, const float* __restrict__ Wkd,
                   const float* __restrict__ Wvd, const float* __restrict__ W1,
                   bf16* __restrict__ ws) {
  const float* src;
  switch (blockIdx.y) {
    case 0: src = Wg;  break; case 1: src = Wqm; break;
    case 2: src = Wkm; break; case 3: src = Wvm; break;
    case 4: src = Wqd; break; case 5: src = Wkd; break;
    case 6: src = Wvd; break; default: src = W1; break;
  }
  bf16* dst = ws + blockIdx.y * (C * C);
  int e = blockIdx.x * 2048 + threadIdx.x * 8;  // dst element index = n*C + k
  int n = e >> 7, k = e & 127;
  v4bf a, b;
#pragma unroll
  for (int i = 0; i < 4; ++i) a[i] = (bf16)src[(k + i) * C + n];
#pragma unroll
  for (int i = 0; i < 4; ++i) b[i] = (bf16)src[(k + 4 + i) * C + n];
  *(v4bf*)(dst + e) = a;
  *(v4bf*)(dst + e + 4) = b;
}

// ---------------- Main fused kernel ----------------
template <bool ASYNCW>
__global__ void __launch_bounds__(TB, 1)
mdi_fused_kernel(const float* __restrict__ em, const float* __restrict__ ed,
                 const float* __restrict__ Wg,
                 const float* __restrict__ Wqm, const float* __restrict__ Wkm,
                 const float* __restrict__ Wvm,
                 const float* __restrict__ Wqd, const float* __restrict__ Wkd,
                 const float* __restrict__ Wvd,
                 const float* __restrict__ W1, const float* __restrict__ W2,
                 const bf16* __restrict__ wsb, float* __restrict__ out) {
  // LDS: 4x32KB bf16 activation tiles + 2x32KB double-buffered weight slots = 192KB.
  __shared__ __align__(64) bf16 sA0[C * C];  // x0_m (em) -> mLA -> node_embed
  __shared__ __align__(64) bf16 sA1[C * C];  // x0_d (ed)
  __shared__ __align__(64) bf16 sA2[C * C];  // x1_m (first 1KB overlays pL coeffs)
  __shared__ __align__(64) bf16 sA3[C * C];  // x1_d (first 512B overlays W2 at end)
  __shared__ __align__(64) bf16 sWa[C * C];  // weight slot 0 (transposed bf16)
  __shared__ __align__(64) bf16 sWb[C * C];  // weight slot 1

  float* pco = (float*)sA2;
  float* w2s = (float*)sA3;

  const int tid = threadIdx.x;
  const int wave = tid >> 5;
  const int lane = tid & 31;
  const int lh = lane & 15;
  const int hi = lane >> 4;
  const int r0 = wave * 16;
  const long base = (long)blockIdx.x * ROWS;

  // Kick off W_gcn -> slot0 under the activation streaming (async path).
  if constexpr (ASYNCW) {
    if (wave == 0) tdm_load_weight(wsb + 0 * C * C, sWa);
  }

  // ---- Phase 0: stream em/ed, row stats -> 2x2 normalized adjacency, bf16 staging
  {
    int row = tid >> 1;
    int ch = (tid & 1) << 6;
    const float* pe = em + (base + row) * C + ch;
    const float* pdd = ed + (base + row) * C + ch;
    float sab = 0.f, sdot = 0.f, smm = 0.f, sdd = 0.f;
#pragma unroll
    for (int c = 0; c < 64; c += 4) {
      v4f a = *(const v4f*)(pe + c);
      v4f b = *(const v4f*)(pdd + c);
#pragma unroll
      for (int u = 0; u < 4; ++u) {
        sab += fabsf(a[u] - b[u]);
        sdot += a[u] * b[u];
        smm += a[u] * a[u];
        sdd += b[u] * b[u];
      }
      v4bf ab = {(bf16)a[0], (bf16)a[1], (bf16)a[2], (bf16)a[3]};
      v4bf bb = {(bf16)b[0], (bf16)b[1], (bf16)b[2], (bf16)b[3]};
      *(v4bf*)(&sA0[row * C + ch + c]) = ab;
      *(v4bf*)(&sA1[row * C + ch + c]) = bb;
    }
    sab += __shfl_xor(sab, 1, 32);
    sdot += __shfl_xor(sdot, 1, 32);
    smm += __shfl_xor(smm, 1, 32);
    sdd += __shfl_xor(sdd, 1, 32);
    if ((tid & 1) == 0) {
      float mlr = sab > 0.f ? sab : 0.01f * sab;                // lrelu(manhattan)
      float cosv = sdot * rsqrtf((smm + FEPS) * (sdd + FEPS));  // cosine sim
      float clr = cosv > 0.f ? cosv : 0.01f * cosv;
      float d3 = 1.f / fmaxf(1.f + mlr, FEPS);
      float d1 = 1.f / fmaxf(1.f + clr, FEPS);
      pco[row * 2 + 0] = fminf(d1, d3);              // pL diagonal
      pco[row * 2 + 1] = fminf(clr * d1, mlr * d3);  // pL off-diagonal
    }
  }
  if constexpr (!ASYNCW) stage_weight(Wg, sWa, tid);
  wait_w<ASYNCW>(wave);  // W_gcn ready; phase-0 LDS visible

  // ---- Phase 1: GCN  x1 = relu(pL @ (x0 @ Wg)) + x0   (weight in slot0)
  issue_w<ASYNCW>(wsb + 1 * C * C, Wqm, sWb, tid, wave);
  float pdv[8], pov[8];
#pragma unroll
  for (int j = 0; j < 8; ++j) {  // read own rows' pL before overwriting sA2
    int rr = (r0 + hi * 8 + j) * 2;
    pdv[j] = pco[rr];
    pov[j] = pco[rr + 1];
  }
#pragma unroll
  for (int t = 0; t < 8; ++t) {
    v8f gm = gemm_tile(sA0, sWa, r0, t * 16, lh, hi);
    v8f gd = gemm_tile(sA1, sWa, r0, t * 16, lh, hi);
#pragma unroll
    for (int j = 0; j < 8; ++j) {
      int idx = (r0 + hi * 8 + j) * C + t * 16 + lh;
      float gmv = gm[j], gdv = gd[j];
      float xm = fmaxf(pdv[j] * gmv + pov[j] * gdv, 0.f) + (float)sA0[idx];
      float xd = fmaxf(pov[j] * gmv + pdv[j] * gdv, 0.f) + (float)sA1[idx];
      sA2[idx] = (bf16)xm;
      sA3[idx] = (bf16)xd;
    }
  }
  wait_w<ASYNCW>(wave);  // Wqm ready in slot1

  // ---- Phase 2: attention (m): Q=slot1, K=slot0, V=slot1; next Wqd -> slot0
  layer_att<false, ASYNCW>(sA0, sA2, sA0, sWb, sWa, sWb,
                           wsb + 2 * C * C, Wkm, wsb + 3 * C * C, Wvm,
                           wsb + 4 * C * C, Wqd, sWa, tid, wave, r0, lh, hi);
  // ---- Phase 3: attention (d), fuse node_embed: Q=slot0, K=slot1, V=slot0; next W1 -> slot1
  layer_att<true, ASYNCW>(sA1, sA3, sA0, sWa, sWb, sWa,
                          wsb + 5 * C * C, Wkd, wsb + 6 * C * C, Wvd,
                          wsb + 7 * C * C, W1, sWb, tid, wave, r0, lh, hi);

  // ---- Phase 4: sigmoid(relu(ne @ W1) @ W2)   (W1 in slot1)
  if (tid < C) w2s[tid] = W2[tid];
  __syncthreads();
  v8f pp = {};
#pragma unroll
  for (int t = 0; t < 8; ++t) {
    v8f h = gemm_tile(sA0, sWb, r0, t * 16, lh, hi);
    float wv = w2s[t * 16 + lh];
#pragma unroll
    for (int j = 0; j < 8; ++j) pp[j] += fmaxf(h[j], 0.f) * wv;
  }
  pp = wave_reduce16(pp);
#pragma unroll
  for (int j = 0; j < 8; ++j) {
    if (lh == j) out[base + r0 + hi * 8 + j] = 1.f / (1.f + __expf(-pp[j]));
  }
}

extern "C" void kernel_launch(void* const* d_in, const int* in_sizes, int n_in,
                              void* d_out, int out_size, void* d_ws, size_t ws_size,
                              hipStream_t stream) {
  (void)n_in; (void)out_size;
  const float* em  = (const float*)d_in[0];
  const float* ed  = (const float*)d_in[1];
  const float* Wg  = (const float*)d_in[2];
  const float* Wqm = (const float*)d_in[3];
  const float* Wkm = (const float*)d_in[4];
  const float* Wvm = (const float*)d_in[5];
  const float* Wqd = (const float*)d_in[6];
  const float* Wkd = (const float*)d_in[7];
  const float* Wvd = (const float*)d_in[8];
  const float* W1  = (const float*)d_in[9];
  const float* W2  = (const float*)d_in[10];
  float* out = (float*)d_out;

  int Bsz = in_sizes[0] / C;  // 262144
  int grid = Bsz / ROWS;      // 2048 workgroups of 8 waves

  const size_t ws_need = (size_t)8 * C * C * sizeof(bf16);  // 256KB of bf16 weights
  if (ws_size >= ws_need) {
    bf16* ws = (bf16*)d_ws;
    cvt_weights_kernel<<<dim3(8, 8), 256, 0, stream>>>(Wg, Wqm, Wkm, Wvm,
                                                       Wqd, Wkd, Wvd, W1, ws);
    mdi_fused_kernel<true><<<dim3(grid), dim3(TB), 0, stream>>>(
        em, ed, Wg, Wqm, Wkm, Wvm, Wqd, Wkd, Wvd, W1, W2, ws, out);
  } else {
    mdi_fused_kernel<false><<<dim3(grid), dim3(TB), 0, stream>>>(
        em, ed, Wg, Wqm, Wkm, Wvm, Wqd, Wkd, Wvd, W1, W2, nullptr, out);
  }
}
